// CrossAttnMem_17394617549128
// MI455X (gfx1250) — matrix operand order repaired and checked by reading, not compiled, not run on hardware
//
#include <hip/hip_runtime.h>
#include <hip/hip_bf16.h>

// MI455X / gfx1250: wave32, WMMA bf16 16x16x32, f32 accumulate.

typedef __attribute__((ext_vector_type(16))) __bf16 v16bf;
typedef __attribute__((ext_vector_type(8)))  float  v8f;

// ---------- bf16 <- f32 : native fptrunc (RNE, lowers to v_cvt_pk_bf16_f32) ----------
__device__ __forceinline__ __bf16 f2bf(float x) { return (__bf16)x; }

// ---------- WMMA wrapper ----------
__device__ __forceinline__ v8f wmma_bf16(v16bf a, v16bf b, v8f c) {
    return __builtin_amdgcn_wmma_f32_16x16x32_bf16(
        /*neg_a=*/false, a, /*neg_b=*/false, b,
        /*c_mod=*/(short)0, c, /*reuse_a=*/false, /*reuse_b=*/false);
}

// ---------- fragment loaders (CDNA5 16-bit layouts, ISA 7.12.2) ----------
// A 16x32 (MxK): lanes 0-15: M=lane, K 0-7 (V0-3) / 16-23 (V4-7);
//                lanes 16-31: M=lane-16, K 8-15 / 24-31.
// Per lane: two contiguous 16B chunks -> 2x global_load_b128.
__device__ __forceinline__ v16bf frag_a_bf16(const __bf16* A, int lda, int lane) {
    int row = lane & 15;
    int kb  = (lane >> 4) << 3;           // 0 or 8
    const __bf16* r = A + (size_t)row * lda + kb;
    union { uint4 q[2]; v16bf f; } u;
    u.q[0] = *(const uint4*)(r);
    u.q[1] = *(const uint4*)(r + 16);
    return u.f;
}

// B 32x16 (KxN): lane holds column N=lane&15; lanes 0-15 K 0-7 / 16-23, lanes 16-31 K 8-15 / 24-31.
// Stored "Bt": B[k][col] = Bt[col*ldb + k] -> two contiguous 16B chunks per lane.
__device__ __forceinline__ v16bf frag_b_bt(const __bf16* Bt, int ldb, int lane) {
    int col = lane & 15;
    int kh  = (lane >> 4) << 3;           // 0 or 8
    const __bf16* r = Bt + (size_t)col * ldb + kh;
    union { uint4 q[2]; v16bf f; } u;
    u.q[0] = *(const uint4*)(r);
    u.q[1] = *(const uint4*)(r + 16);
    return u.f;
}

// ======================================================================
// Kernel 0a: bulk f32 -> bf16 convert (8 elements / thread, vectorized)
// ======================================================================
__global__ __launch_bounds__(256) void cvt_bf16_kernel(
    const float* __restrict__ src, __bf16* __restrict__ dst, int n8)
{
    int i = blockIdx.x * 256 + threadIdx.x;
    if (i >= n8) return;
    const float4* s = (const float4*)src + (size_t)i * 2;
    float4 a = s[0], b = s[1];
    union { uint4 q; __bf16 h[8]; } u;
    u.h[0] = f2bf(a.x); u.h[1] = f2bf(a.y); u.h[2] = f2bf(a.z); u.h[3] = f2bf(a.w);
    u.h[4] = f2bf(b.x); u.h[5] = f2bf(b.y); u.h[6] = f2bf(b.z); u.h[7] = f2bf(b.w);
    ((uint4*)dst)[i] = u.q;
}

// ======================================================================
// Kernel 0b: weight repack/transpose  W[K,N] f32 -> Wt[N,K] bf16 (Bt layout)
// ======================================================================
__global__ __launch_bounds__(256) void wt_repack_kernel(
    const float* __restrict__ W, __bf16* __restrict__ Wt, int K, int N)
{
    int i = blockIdx.x * 256 + threadIdx.x;
    if (i >= K * N) return;
    int n = i / K, k = i - n * K;
    Wt[(size_t)n * K + k] = f2bf(W[(size_t)k * N + n]);
}

// ======================================================================
// Kernel 1: projections  C = embB @ Wt  (M=16384,K=64,N=512), grid.z = 3
//   A: bf16 [16384,64];  Wt: bf16 Bt layout [512 cols][64 k]
//   mode 0: store normal   [16384, 512]  bf16
//   mode 1: store transposed [b, hc, n]  bf16  (b = row>>12, n = row&4095)
// ======================================================================
__global__ __launch_bounds__(256) void proj_kernel(
    const __bf16* a0, const __bf16* a1, const __bf16* a2,
    const __bf16* wt0, const __bf16* wt1, const __bf16* wt2,
    __bf16* o0, __bf16* o1, __bf16* o2,
    int m0, int m1, int m2)
{
    int z = blockIdx.z;
    const __bf16* A  = (z == 0) ? a0  : (z == 1) ? a1  : a2;
    const __bf16* Wt = (z == 0) ? wt0 : (z == 1) ? wt1 : wt2;
    __bf16*       O  = (z == 0) ? o0  : (z == 1) ? o1  : o2;
    int        mode  = (z == 0) ? m0  : (z == 1) ? m1  : m2;

    int lane  = threadIdx.x & 31;
    int wid   = threadIdx.x >> 5;     // 0..7 -> N strip of 64
    int Mbase = blockIdx.x * 16;      // 1024 M tiles
    int nb    = wid * 64;

    v8f acc[4];
#pragma unroll
    for (int t = 0; t < 4; ++t)
#pragma unroll
        for (int j = 0; j < 8; ++j) acc[t][j] = 0.f;

#pragma unroll
    for (int ks = 0; ks < 2; ++ks) {
        int k0 = ks * 32;
        v16bf af = frag_a_bf16(A + (size_t)Mbase * 64 + k0, 64, lane);
        v16bf bfr[4];
#pragma unroll
        for (int t = 0; t < 4; ++t)
            bfr[t] = frag_b_bt(Wt + (size_t)(nb + t * 16) * 64 + k0, 64, lane);
#pragma unroll
        for (int t = 0; t < 4; ++t)
            acc[t] = wmma_bf16(af, bfr[t], acc[t]);
    }

    int colL   = lane & 15;
    int rhalf  = (lane >> 4) * 8;
    if (mode == 0) {
#pragma unroll
        for (int t = 0; t < 4; ++t)
#pragma unroll
            for (int j = 0; j < 8; ++j) {
                int row = Mbase + rhalf + j;
                O[(size_t)row * 512 + nb + t * 16 + colL] = f2bf(acc[t][j]);
            }
    } else {
#pragma unroll
        for (int t = 0; t < 4; ++t)
#pragma unroll
            for (int j = 0; j < 8; ++j) {
                int row = Mbase + rhalf + j;
                int b = row >> 12, n = row & 4095;
                int hc = nb + t * 16 + colL;
                O[((size_t)b * 512 + hc) * 4096 + n] = f2bf(acc[t][j]);
            }
    }
}

// ======================================================================
// Kernel 2: self-attn scores [64x64] over K=4096 + InstanceNorm + softmax
//   one block (128 thr = 4 waves) per (b,h); fused in LDS.
//   Qt, Kt: [B, HC, N] bf16 (transposed).  attn: [32, 64, 64] bf16.
// ======================================================================
__global__ __launch_bounds__(128) void scores_softmax_kernel(
    const __bf16* Qt, const __bf16* Kt, __bf16* attn)
{
    __shared__ float S[64 * 64];
    __shared__ float redA[128], redB[128];

    int bh   = blockIdx.x;                        // 0..31
    int hc0  = (bh >> 3) * 512 + (bh & 7) * 64;   // b*HC + h*64
    int tid  = threadIdx.x;
    int lane = tid & 31;
    int wid  = tid >> 5;                          // c-tile 0..3

    const __bf16* Ab = Qt + (size_t)(hc0 + wid * 16) * 4096;

    v8f acc[4];
#pragma unroll
    for (int t = 0; t < 4; ++t)
#pragma unroll
        for (int j = 0; j < 8; ++j) acc[t][j] = 0.f;

#pragma unroll 2
    for (int ks = 0; ks < 128; ++ks) {
        int k0 = ks * 32;
        __builtin_prefetch(Ab + k0 + 512, 0, 1);   // global_prefetch_b8
        v16bf af = frag_a_bf16(Ab + k0, 4096, lane);
        v16bf bfr[4];
#pragma unroll
        for (int t = 0; t < 4; ++t)
            bfr[t] = frag_b_bt(Kt + (size_t)(hc0 + t * 16) * 4096 + k0, 4096, lane);
#pragma unroll
        for (int t = 0; t < 4; ++t)
            acc[t] = wmma_bf16(af, bfr[t], acc[t]);
    }

    int colL  = lane & 15;
    int rhalf = (lane >> 4) * 8;
#pragma unroll
    for (int t = 0; t < 4; ++t)
#pragma unroll
        for (int j = 0; j < 8; ++j)
            S[(wid * 16 + rhalf + j) * 64 + t * 16 + colL] = acc[t][j];
    __syncthreads();

    // InstanceNorm stats over all 64x64
    float s = 0.f, s2 = 0.f;
    for (int i = tid; i < 4096; i += 128) { float v = S[i]; s += v; s2 += v * v; }
    redA[tid] = s; redB[tid] = s2; __syncthreads();
    for (int off = 64; off; off >>= 1) {
        if (tid < off) { redA[tid] += redA[tid + off]; redB[tid] += redB[tid + off]; }
        __syncthreads();
    }
    float mean = redA[0] * (1.f / 4096.f);
    float var  = redB[0] * (1.f / 4096.f) - mean * mean;
    float rstd = rsqrtf(var + 1e-5f);
    __syncthreads();

    // per-row softmax (row c handled by thread c)
    if (tid < 64) {
        float mx = -3.4e38f;
        for (int d = 0; d < 64; ++d) {
            float v = (S[tid * 64 + d] - mean) * rstd;
            mx = fmaxf(mx, v);
        }
        float sum = 0.f;
        for (int d = 0; d < 64; ++d) {
            float e = __expf((S[tid * 64 + d] - mean) * rstd - mx);
            S[tid * 64 + d] = e; sum += e;
        }
        float inv = 1.f / sum;
        for (int d = 0; d < 64; ++d)
            attn[((size_t)bh * 64 + tid) * 64 + d] = f2bf(S[tid * 64 + d] * inv);
    }
}

// ======================================================================
// Kernel 3: self ctx  ctx[b,h,c,n] = sum_d attn[c,d] * v[n,d]
//   per wave: 64(c) x 16(n); write ctx_flat[b,n, c*8+h] bf16.
// ======================================================================
__global__ __launch_bounds__(256) void ctx_kernel(
    const __bf16* attn, const __bf16* Vn, __bf16* ctx)
{
    int job  = blockIdx.x * 8 + (threadIdx.x >> 5);  // 0..8191
    int lane = threadIdx.x & 31;
    int bh   = job >> 8;
    int n0   = (job & 255) * 16;
    int b = bh >> 3, h = bh & 7;

    const __bf16* Ab = attn + (size_t)bh * 64 * 64;
    const __bf16* Bt = Vn + ((size_t)b * 4096 + n0) * 512 + h * 64;  // Bt[col*512 + d]

    v8f acc[4];
#pragma unroll
    for (int t = 0; t < 4; ++t)
#pragma unroll
        for (int j = 0; j < 8; ++j) acc[t][j] = 0.f;

#pragma unroll
    for (int ks = 0; ks < 2; ++ks) {
        int k0 = ks * 32;
        v16bf bf = frag_b_bt(Bt + k0, 512, lane);
        v16bf afr[4];
#pragma unroll
        for (int t = 0; t < 4; ++t)
            afr[t] = frag_a_bf16(Ab + (size_t)(t * 16) * 64 + k0, 64, lane);
#pragma unroll
        for (int t = 0; t < 4; ++t)
            acc[t] = wmma_bf16(afr[t], bf, acc[t]);
    }

    int colL  = lane & 15;
    int rhalf = (lane >> 4) * 8;
#pragma unroll
    for (int t = 0; t < 4; ++t)
#pragma unroll
        for (int j = 0; j < 8; ++j) {
            int c = t * 16 + rhalf + j;
            ctx[((size_t)b * 4096 + n0 + colL) * 512 + c * 8 + h] = f2bf(acc[t][j]);
        }
}

// ======================================================================
// Kernel 4: output GEMM  [16384,512]bf16 @ Wt(bf16 Bt [64][512]) -> f32
// ======================================================================
__global__ __launch_bounds__(256) void out_gemm_kernel(
    const __bf16* A, const __bf16* Wto, float* out)
{
    int Mtile = blockIdx.x * 8 + (threadIdx.x >> 5);  // 0..1023
    int lane  = threadIdx.x & 31;
    int Mbase = Mtile * 16;

    v8f acc[4];
#pragma unroll
    for (int t = 0; t < 4; ++t)
#pragma unroll
        for (int j = 0; j < 8; ++j) acc[t][j] = 0.f;

#pragma unroll 2
    for (int ks = 0; ks < 16; ++ks) {
        int k0 = ks * 32;
        v16bf af = frag_a_bf16(A + (size_t)Mbase * 512 + k0, 512, lane);
        v16bf bfr[4];
#pragma unroll
        for (int t = 0; t < 4; ++t)
            bfr[t] = frag_b_bt(Wto + (size_t)(t * 16) * 512 + k0, 512, lane);
#pragma unroll
        for (int t = 0; t < 4; ++t)
            acc[t] = wmma_bf16(af, bfr[t], acc[t]);
    }

    int colL  = lane & 15;
    int rhalf = (lane >> 4) * 8;
#pragma unroll
    for (int t = 0; t < 4; ++t)
#pragma unroll
        for (int j = 0; j < 8; ++j)
            out[(size_t)(Mbase + rhalf + j) * 64 + t * 16 + colL] = acc[t][j];
}

// ======================================================================
// Kernel 5: cross scores  cs[b,d,m] = sum_n QLt[b,d,n] * KCt[m,n]
//   per wave: 16(d) x 64(m), K=4096.   cs: [4, 512, 2048] f32
// ======================================================================
__global__ __launch_bounds__(256) void cscores_kernel(
    const __bf16* QLt, const __bf16* KCt, float* cs)
{
    int job  = blockIdx.x * 8 + (threadIdx.x >> 5);  // 0..4095
    int lane = threadIdx.x & 31;
    int b     = job >> 10;
    int r     = job & 1023;
    int Mtile = r >> 5;       // d tile 0..31
    int Ns    = r & 31;       // m strip of 64

    const __bf16* Ab = QLt + ((size_t)b * 512 + Mtile * 16) * 4096;

    v8f acc[4];
#pragma unroll
    for (int t = 0; t < 4; ++t)
#pragma unroll
        for (int j = 0; j < 8; ++j) acc[t][j] = 0.f;

#pragma unroll 2
    for (int ks = 0; ks < 128; ++ks) {
        int k0 = ks * 32;
        __builtin_prefetch(Ab + k0 + 512, 0, 1);
        v16bf af = frag_a_bf16(Ab + k0, 4096, lane);
        v16bf bfr[4];
#pragma unroll
        for (int t = 0; t < 4; ++t)
            bfr[t] = frag_b_bt(KCt + (size_t)(Ns * 64 + t * 16) * 4096 + k0, 4096, lane);
#pragma unroll
        for (int t = 0; t < 4; ++t)
            acc[t] = wmma_bf16(af, bfr[t], acc[t]);
    }

    int colL  = lane & 15;
    int rhalf = (lane >> 4) * 8;
#pragma unroll
    for (int t = 0; t < 4; ++t)
#pragma unroll
        for (int j = 0; j < 8; ++j)
            cs[((size_t)b * 512 + Mtile * 16 + rhalf + j) * 2048 + Ns * 64 + t * 16 + colL]
                = acc[t][j];
}

// ======================================================================
// Kernel 6: cross InstanceNorm stats per b over [512,2048]
// ======================================================================
__global__ __launch_bounds__(256) void cstats_kernel(const float* cs, float* stats)
{
    __shared__ float rA[256], rB[256];
    int b = blockIdx.x, tid = threadIdx.x;
    const float* p = cs + (size_t)b * 512 * 2048;
    float s = 0.f, s2 = 0.f;
    for (size_t i = tid; i < (size_t)512 * 2048; i += 256) { float v = p[i]; s += v; s2 += v * v; }
    rA[tid] = s; rB[tid] = s2; __syncthreads();
    for (int off = 128; off; off >>= 1) {
        if (tid < off) { rA[tid] += rA[tid + off]; rB[tid] += rB[tid + off]; }
        __syncthreads();
    }
    if (tid == 0) {
        float inv = 1.f / (512.f * 2048.f);
        float m = rA[0] * inv;
        float var = rB[0] * inv - m * m;
        stats[b * 2]     = m;
        stats[b * 2 + 1] = rsqrtf(var + 1e-5f);
    }
}

// ======================================================================
// Kernel 7: cross softmax per row (2048 cols) -> bf16 cattn
// ======================================================================
__global__ __launch_bounds__(256) void csoftmax_kernel(
    const float* cs, const float* stats, __bf16* ca)
{
    __shared__ float red[256];
    int row = blockIdx.x;           // 0..2047 (b*512 + d)
    int b = row >> 9, tid = threadIdx.x;
    float mean = stats[b * 2], rstd = stats[b * 2 + 1];
    const float* p = cs + (size_t)row * 2048;

    float vals[8];
    float mx = -3.4e38f;
#pragma unroll
    for (int i = 0; i < 8; ++i) {
        vals[i] = (p[tid + i * 256] - mean) * rstd;
        mx = fmaxf(mx, vals[i]);
    }
    red[tid] = mx; __syncthreads();
    for (int off = 128; off; off >>= 1) {
        if (tid < off) red[tid] = fmaxf(red[tid], red[tid + off]);
        __syncthreads();
    }
    mx = red[0]; __syncthreads();

    float s = 0.f;
#pragma unroll
    for (int i = 0; i < 8; ++i) { vals[i] = __expf(vals[i] - mx); s += vals[i]; }
    red[tid] = s; __syncthreads();
    for (int off = 128; off; off >>= 1) {
        if (tid < off) red[tid] += red[tid + off];
        __syncthreads();
    }
    float inv = 1.f / red[0];

    __bf16* o = ca + (size_t)row * 2048;
#pragma unroll
    for (int i = 0; i < 8; ++i) o[tid + i * 256] = f2bf(vals[i] * inv);
}

// ======================================================================
// Kernel 8: cross ctx  cctx[b,d,n] = sum_m cattn[b,d,m] * vc[n,m]
//   vc[n, b2*512+e] = VCn[b2, n, e].  per wave: 16(d) x 64(n), K=2048.
//   write cctx_flat[b, n, d] bf16.
// ======================================================================
__global__ __launch_bounds__(256) void cctx_kernel(
    const __bf16* ca, const __bf16* VCn, __bf16* cctx)
{
    int job  = blockIdx.x * 8 + (threadIdx.x >> 5);  // 0..8191
    int lane = threadIdx.x & 31;
    int b     = job >> 11;
    int r     = job & 2047;
    int Mtile = r >> 6;       // d tile 0..31
    int n0    = (r & 63) * 64;

    const __bf16* Ab = ca + ((size_t)b * 512 + Mtile * 16) * 2048;

    v8f acc[4];
#pragma unroll
    for (int t = 0; t < 4; ++t)
#pragma unroll
        for (int j = 0; j < 8; ++j) acc[t][j] = 0.f;

#pragma unroll 2
    for (int ks = 0; ks < 64; ++ks) {
        int k0 = ks * 32;
        int b2 = k0 >> 9, e0 = k0 & 511;        // 32-wide k-step never crosses b2
        v16bf af = frag_a_bf16(Ab + k0, 2048, lane);
        v16bf bfr[4];
#pragma unroll
        for (int t = 0; t < 4; ++t)
            bfr[t] = frag_b_bt(VCn + ((size_t)b2 * 4096 + n0 + t * 16) * 512 + e0, 512, lane);
#pragma unroll
        for (int t = 0; t < 4; ++t)
            acc[t] = wmma_bf16(af, bfr[t], acc[t]);
    }

    int colL  = lane & 15;
    int rhalf = (lane >> 4) * 8;
#pragma unroll
    for (int t = 0; t < 4; ++t)
#pragma unroll
        for (int j = 0; j < 8; ++j) {
            int n = n0 + t * 16 + colL;
            int d = Mtile * 16 + rhalf + j;
            cctx[((size_t)b * 4096 + n) * 512 + d] = f2bf(acc[t][j]);
        }
}

// ======================================================================
// Host launcher
// ======================================================================
extern "C" void kernel_launch(void* const* d_in, const int* in_sizes, int n_in,
                              void* d_out, int out_size, void* d_ws, size_t ws_size,
                              hipStream_t stream)
{
    (void)in_sizes; (void)n_in; (void)out_size; (void)ws_size;

    const float* emb       = (const float*)d_in[0];        // [8,4096,64]
    const float* W_qu      = (const float*)d_in[3];
    const float* W_ku      = (const float*)d_in[4];
    const float* W_vu      = (const float*)d_in[5];
    const float* W_ql2u    = (const float*)d_in[6];
    const float* W_kl2u    = (const float*)d_in[7];
    const float* W_vl2u    = (const float*)d_in[8];
    const float* W_out_u   = (const float*)d_in[9];
    const float* W_out_l2u = (const float*)d_in[10];

    float* outF = (float*)d_out;                            // [8,4096,64]

    char* ws = (char*)d_ws;
    __bf16* PA    = (__bf16*)(ws + 0);                       // 16 MB : Qt / QLt  [B,HC,N]
    __bf16* PB    = (__bf16*)(ws + (16ull << 20));           // 16 MB : Kt / KCt  [B,HC,N]
    __bf16* PC    = (__bf16*)(ws + (32ull << 20));           // 16 MB : Vn / VCn  [B,N,HC]
    __bf16* ATT   = (__bf16*)(ws + (48ull << 20));           // 256 KB: self attn [32,64,64]
    float*  STATS = (float*) (ws + (48ull << 20) + (512ull << 10)); // 8 floats
    __bf16* CTX   = (__bf16*)(ws + (49ull << 20));           // 16 MB : ctx / cctx flat
    float*  CS    = (float*) (ws + (65ull << 20));           // 16 MB : cross scores f32
    __bf16* CA    = (__bf16*)(ws + (81ull << 20));           // 8 MB  : cross attn bf16
    __bf16* EMBB  = (__bf16*)(ws + (89ull << 20));           // 4 MB  : emb bf16 [8,4096,64]
    __bf16* WTB   = (__bf16*)(ws + (93ull << 20));           // 8 x 64KB : repacked weights

    __bf16* Wt_qu    = WTB + 0 * 32768;
    __bf16* Wt_ku    = WTB + 1 * 32768;
    __bf16* Wt_vu    = WTB + 2 * 32768;
    __bf16* Wt_ql2u  = WTB + 3 * 32768;
    __bf16* Wt_kl2u  = WTB + 4 * 32768;
    __bf16* Wt_vl2u  = WTB + 5 * 32768;
    __bf16* Wt_out_u = WTB + 6 * 32768;
    __bf16* Wt_out_l = WTB + 7 * 32768;

    const __bf16* embB_l = EMBB;
    const __bf16* embB_u = EMBB + (size_t)4 * 4096 * 64;

    // ---- preprocessing: emb -> bf16, weights -> bf16 Bt layout ----
    cvt_bf16_kernel<<<1024, 256, 0, stream>>>(emb, EMBB, (8 * 4096 * 64) / 8);
    wt_repack_kernel<<<128, 256, 0, stream>>>(W_qu,      Wt_qu,    64, 512);
    wt_repack_kernel<<<128, 256, 0, stream>>>(W_ku,      Wt_ku,    64, 512);
    wt_repack_kernel<<<128, 256, 0, stream>>>(W_vu,      Wt_vu,    64, 512);
    wt_repack_kernel<<<128, 256, 0, stream>>>(W_ql2u,    Wt_ql2u,  64, 512);
    wt_repack_kernel<<<128, 256, 0, stream>>>(W_kl2u,    Wt_kl2u,  64, 512);
    wt_repack_kernel<<<128, 256, 0, stream>>>(W_vl2u,    Wt_vl2u,  64, 512);
    wt_repack_kernel<<<128, 256, 0, stream>>>(W_out_u,   Wt_out_u, 512, 64);
    wt_repack_kernel<<<128, 256, 0, stream>>>(W_out_l2u, Wt_out_l, 512, 64);

    // ---- self-attention path ----
    proj_kernel<<<dim3(1024, 1, 3), 256, 0, stream>>>(
        embB_u, embB_u, embB_u, Wt_qu, Wt_ku, Wt_vu, PA, PB, PC, 1, 1, 0);
    scores_softmax_kernel<<<32, 128, 0, stream>>>(PA, PB, ATT);
    ctx_kernel<<<1024, 256, 0, stream>>>(ATT, PC, CTX);
    out_gemm_kernel<<<128, 256, 0, stream>>>(CTX, Wt_out_u, outF + (size_t)4 * 4096 * 64);

    // ---- cross-attention path (reuses PA/PB/PC/CTX) ----
    proj_kernel<<<dim3(1024, 1, 3), 256, 0, stream>>>(
        embB_l, embB_u, embB_u, Wt_ql2u, Wt_kl2u, Wt_vl2u, PA, PB, PC, 1, 1, 0);
    cscores_kernel<<<512, 256, 0, stream>>>(PA, PB, CS);
    cstats_kernel<<<4, 256, 0, stream>>>(CS, STATS);
    csoftmax_kernel<<<2048, 256, 0, stream>>>(CS, STATS, CA);
    cctx_kernel<<<1024, 256, 0, stream>>>(CA, PC, CTX);
    out_gemm_kernel<<<128, 256, 0, stream>>>(CTX, Wt_out_l, outF);
}
